// LinearEquivariant_9560597201465
// MI455X (gfx1250) — matrix operand order repaired and checked by reading, not compiled, not run on hardware
//
#include <hip/hip_runtime.h>
#include <hip/hip_bf16.h>

// ---------------------------------------------------------------------------
// out[b,n,o] = sum_i x[b,n,i]*alpha[i,o]  +  r[b,o]
//   r[b,o]   = sum_i x_sum[b,i]*beta[i,o] + sum_i bias[i,o]
//
// M = 8*16384 = 131072 flattened rows, K = 128, N = 128.
// Memory-bound. With NT output stores, x (64 MB) stays resident in the 192 MB
// L2 after pass 1, so HBM traffic ~ 64 MB read + 64 MB write (~5.5 us floor).
// GEMM via bf16 split precision:
//   x = xh + xl, alpha = ah + al  (bf16 each)
//   x@alpha ~= xh@ah + xl@ah + xh@al        (3x v_wmma_f32_16x16x32_bf16)
// fp32 accumulate; error ~2^-16 relative, fp32-class for K=128 dots.
// ---------------------------------------------------------------------------

typedef __attribute__((ext_vector_type(16))) __bf16 v16bf;
typedef __attribute__((ext_vector_type(8)))  float  v8f;

#define B_DIM   8
#define N_DIM   16384
#define C_DIM   128
#define M_DIM   (B_DIM * N_DIM)      // 131072 rows
#define ROWT    8                    // 16-row tiles per block -> 128 rows/block
#define NCHUNK  128                  // column-sum chunks per batch (128 rows each)
#define LDS_STRIDE 68                // uints per row: 136 halves (128 + 8 pad)

// packed fp32x2 -> bf16x2 (RNE) via the gfx12-family VOP3 packed convert.
// (clang's __builtin_amdgcn_cvt_pk_bf16_f32 is not exposed for gfx1250 on this
// toolchain, so emit the instruction directly.)
__device__ __forceinline__ unsigned int pk_bf16(float a0, float a1) {
    unsigned int u;
    asm("v_cvt_pk_bf16_f32 %0, %1, %2" : "=v"(u) : "v"(a0), "v"(a1));
    return u;
}

// split 8 floats into bf16 hi/lo pairs and store both LDS fragments
__device__ __forceinline__ void cvt_store(unsigned int* __restrict__ lh,
                                          unsigned int* __restrict__ ll,
                                          int idx, float4 f0, float4 f1) {
    float fv[8] = { f0.x, f0.y, f0.z, f0.w, f1.x, f1.y, f1.z, f1.w };
    unsigned int uh[4], ul[4];
#pragma unroll
    for (int j = 0; j < 4; ++j) {
        float a0 = fv[2 * j], a1 = fv[2 * j + 1];
        unsigned int h = pk_bf16(a0, a1);
        float fh0 = __uint_as_float(h << 16);
        float fh1 = __uint_as_float(h & 0xFFFF0000u);
        uh[j] = h;
        ul[j] = pk_bf16(a0 - fh0, a1 - fh1);
    }
    uint4 qh; qh.x = uh[0]; qh.y = uh[1]; qh.z = uh[2]; qh.w = uh[3];
    uint4 ql; ql.x = ul[0]; ql.y = ul[1]; ql.z = ul[2]; ql.w = ul[3];
    *reinterpret_cast<uint4*>(lh + idx) = qh;
    *reinterpret_cast<uint4*>(ll + idx) = ql;
}

// ---------------------------------------------------------------------------
// Pass 1: partial column sums of x. Deterministic (no fp atomics).
// Regular-temporal loads pull x into L2 where the GEMM pass re-reads it.
// ---------------------------------------------------------------------------
__global__ __launch_bounds__(128)
void le_colsum_partial(const float* __restrict__ x, float* __restrict__ partial) {
    const int b  = blockIdx.x >> 7;          // / NCHUNK
    const int ch = blockIdx.x & 127;
    const int c  = threadIdx.x;
    const float* p = x + ((size_t)b * N_DIM + (size_t)ch * 128) * C_DIM + c;
    float s = 0.0f;
#pragma unroll 4
    for (int r = 0; r < 128; ++r)
        s += p[(size_t)r * C_DIM];
    __builtin_nontemporal_store(s, &partial[((size_t)blockIdx.x) * 128 + c]);
}

// ---------------------------------------------------------------------------
// Pass 2: r[b,o] = x_sum[b,:] @ beta[:,o] + colsum(bias)[o].   grid=B, 128 thr.
// ---------------------------------------------------------------------------
__global__ __launch_bounds__(128)
void le_make_r(const float* __restrict__ partial,
               const float* __restrict__ beta,
               const float* __restrict__ bias,
               float* __restrict__ rvec) {
    const int b = blockIdx.x;
    const int t = threadIdx.x;
    __shared__ float xs[C_DIM];
    const float* pp = partial + (size_t)b * NCHUNK * 128 + t;
    float s = 0.0f;
#pragma unroll 4
    for (int ch = 0; ch < NCHUNK; ++ch)
        s += pp[(size_t)ch * 128];
    xs[t] = s;
    __syncthreads();
    float acc = 0.0f;
#pragma unroll 4
    for (int i = 0; i < C_DIM; ++i)
        acc += xs[i] * beta[i * C_DIM + t] + bias[i * C_DIM + t];
    rvec[b * C_DIM + t] = acc;
}

// ---------------------------------------------------------------------------
// Pass 3: main GEMM. 256 threads = 8 waves/block. Wave w owns column tile w
// (16 output columns) with alpha hi/lo fragments resident in VGPRs.
// Double-buffered LDS staging: one barrier per 16-row tile; next tile's
// global loads issue before the current tile's WMMA chain. Output stores are
// non-temporal so x stays L2-resident.
// ---------------------------------------------------------------------------
__global__ __launch_bounds__(256)
void le_gemm_bf16x3(const float* __restrict__ x,
                    const float* __restrict__ alpha,
                    const float* __restrict__ rvec,
                    float* __restrict__ out) {
    __shared__ unsigned int lds_hi[2][16 * LDS_STRIDE];
    __shared__ unsigned int lds_lo[2][16 * LDS_STRIDE];

    const int tid  = threadIdx.x;
    const int lane = tid & 31;
    const int wave = tid >> 5;              // 0..7 -> column tile
    const int nloc = lane & 15;
    const int ocol = wave * 16 + nloc;      // absolute output column

    // ---- preload B fragments (alpha hi/lo) for 4 k-steps, resident in VGPRs.
    // B layout (K x 16, bf16): lane l holds column l%16; halves h=0..15 carry
    // K = kt*32 + (l<16 ? 0:16) + h.
    v16bf bh[4], bl[4];
    const int kb2 = (lane < 16) ? 0 : 16;
#pragma unroll
    for (int kt = 0; kt < 4; ++kt) {
        union { v16bf v; unsigned int u[8]; } H, L;
#pragma unroll
        for (int j = 0; j < 8; ++j) {
            float f0 = alpha[(kt * 32 + kb2 + 2 * j) * C_DIM + ocol];
            float f1 = alpha[(kt * 32 + kb2 + 2 * j + 1) * C_DIM + ocol];
            unsigned int h = pk_bf16(f0, f1);
            float fh0 = __uint_as_float(h << 16);
            float fh1 = __uint_as_float(h & 0xFFFF0000u);
            H.u[j] = h;
            L.u[j] = pk_bf16(f0 - fh0, f1 - fh1);
        }
        bh[kt] = H.v;
        bl[kt] = L.v;
    }

    const size_t row0 = (size_t)blockIdx.x * (16 * ROWT);
    const int    bidx = (int)(row0 >> 14);         // row0 / 16384 (block-uniform)
    const float  rval = rvec[bidx * C_DIM + ocol];

    // staging role: 8 consecutive floats of one row
    const int srow = tid >> 4;                     // 0..15
    const int scol = (tid & 15) * 8;               // 0..120
    const int sidx = srow * LDS_STRIDE + (scol >> 1);
    const int kb   = (lane < 16) ? 0 : 16;         // A-fragment K sub-base
    const int abase = nloc * LDS_STRIDE + (kb >> 1);

    // ---- prologue: stage tile 0 into buffer 0
    {
        const float* src = x + (row0 + srow) * C_DIM + scol;
        float4 f0 = *reinterpret_cast<const float4*>(src);
        float4 f1 = *reinterpret_cast<const float4*>(src + 4);
        cvt_store(lds_hi[0], lds_lo[0], sidx, f0, f1);
    }

    for (int rt = 0; rt < ROWT; ++rt) {
        __syncthreads();   // buffer rt&1 ready; previous buffer fully consumed
        const int cur = rt & 1;
        const size_t row_start = row0 + (size_t)rt * 16;

        // ---- issue next tile's global loads first (overlap with WMMAs)
        float4 g0, g1;
        const bool have_next = (rt + 1 < ROWT);
        if (have_next) {
            const float* src = x + (row_start + 16 + srow) * C_DIM + scol;
            g0 = *reinterpret_cast<const float4*>(src);
            g1 = *reinterpret_cast<const float4*>(src + 4);
        }

        // ---- compute: acc starts at r[b,o] (adds the broadcast term once)
        v8f acc = { rval, rval, rval, rval, rval, rval, rval, rval };
        const unsigned int* lh = lds_hi[cur];
        const unsigned int* ll = lds_lo[cur];
#pragma unroll
        for (int kt = 0; kt < 4; ++kt) {
            // A layout (16 x 32, bf16): lane l holds row l%16; halves 0..15 = K
            // kt*32+kb..+15 (kb in {0,16}).
            const int base = abase + kt * 16;
            union { v16bf v; uint4 q[2]; } AH, AL;
            AH.q[0] = *reinterpret_cast<const uint4*>(lh + base);
            AH.q[1] = *reinterpret_cast<const uint4*>(lh + base + 4);
            AL.q[0] = *reinterpret_cast<const uint4*>(ll + base);
            AL.q[1] = *reinterpret_cast<const uint4*>(ll + base + 4);

            acc = __builtin_amdgcn_wmma_f32_16x16x32_bf16(
                      false, AH.v, false, bh[kt], (short)0, acc, false, false);
            acc = __builtin_amdgcn_wmma_f32_16x16x32_bf16(
                      false, AL.v, false, bh[kt], (short)0, acc, false, false);
            acc = __builtin_amdgcn_wmma_f32_16x16x32_bf16(
                      false, AH.v, false, bl[kt], (short)0, acc, false, false);
        }

        // ---- stage next tile into the other buffer (no barrier needed here:
        // last readers of that buffer passed the barrier above)
        if (have_next)
            cvt_store(lds_hi[cur ^ 1], lds_lo[cur ^ 1], sidx, g0, g1);

        // ---- epilogue: C/D layout: VGPR v -> row (lane<16 ? v : v+8), col o.
        // Non-temporal: out is never re-read; keep x resident in L2.
        float* orow = out + (row_start + ((lane < 16) ? 0 : 8)) * C_DIM + ocol;
#pragma unroll
        for (int v = 0; v < 8; ++v)
            __builtin_nontemporal_store(acc[v], &orow[(size_t)v * C_DIM]);
    }
}

// ---------------------------------------------------------------------------
// Launch. ws layout: [0, 131072) floats: column-sum partials
//                    [131072, 132096) floats: r[b,o]
// ---------------------------------------------------------------------------
extern "C" void kernel_launch(void* const* d_in, const int* in_sizes, int n_in,
                              void* d_out, int out_size, void* d_ws, size_t ws_size,
                              hipStream_t stream) {
    const float* x     = (const float*)d_in[0];
    const float* alpha = (const float*)d_in[1];
    const float* beta  = (const float*)d_in[2];
    const float* bias  = (const float*)d_in[3];
    float*       out   = (float*)d_out;

    float* partial = (float*)d_ws;                      // 8*128*128 floats
    float* rvec    = partial + (size_t)B_DIM * NCHUNK * 128;

    le_colsum_partial<<<B_DIM * NCHUNK, 128, 0, stream>>>(x, partial);
    le_make_r<<<B_DIM, 128, 0, stream>>>(partial, beta, bias, rvec);
    le_gemm_bf16x3<<<M_DIM / (16 * ROWT), 256, 0, stream>>>(x, alpha, rvec, out);
}
